// SymSimGCNNet_9216999817949
// MI455X (gfx1250) — compile-verified, matrix operand order checked by reading, not run on hardware
//
#include <hip/hip_runtime.h>
#include <hip/hip_bf16.h>

// SymSimGCNNet fused for MI455X (gfx1250, wave32, WMMA bf16 16x16x32).
//
// Shapes (fixed by reference): N=62 nodes, B=512 graphs, F_IN=128, H=64, C=3, K=2.
// All graphs share one symmetric adjacency -> precompute A2 = (D^-1/2 W D^-1/2)^2 once,
// then per graph:  t = A2 @ x_g ; h = relu(t @ W1^T + b1) ; pooled = colsum_{m<62}(h);
// logits = pooled @ W2^T + b2 ; logp = log_softmax(logits).
//
// All WMMA operands are stored K-contiguous per lane (B operands kept transposed:
// x^T and W1 in natural [n][k] layout), so every fragment load is 2x ds_load_b128.
// Row strides padded (64->72, 128->136 bf16) for conflict-free LDS b128 loads.
// x staging processes node-row pairs so bf16 conversion uses packed cvt + b32 stores.

typedef __attribute__((ext_vector_type(16))) __bf16 v16bf;
typedef __attribute__((ext_vector_type(8)))  __bf16 v8bf;
typedef __attribute__((ext_vector_type(8)))  float  v8f;

#define NB    512
#define NN    62
#define FIN   128
#define HH    64
#define S64   72    // padded stride for K=64 rows  (144 B = 36 banks -> conflict-free)
#define S128  136   // padded stride for K=128 rows (272 B = 68 banks -> conflict-free)

// Fragment load: 16 bf16 along K for this lane's row (A: row m, B: column n).
// lane l: kb = (l>>4)*8; elements K = kb..kb+7 and kb+16..kb+23 of the 32-chunk.
__device__ inline v16bf frag_row(const __bf16* p /* = row base + kc*32 */, int lane) {
    const int kb = (lane >> 4) << 3;
    v8bf lo = *(const v8bf*)(p + kb);        // ds_load_b128
    v8bf hi = *(const v8bf*)(p + kb + 16);   // ds_load_b128
    return __builtin_shufflevector(lo, hi, 0, 1, 2, 3, 4, 5, 6, 7,
                                   8, 9, 10, 11, 12, 13, 14, 15);
}

__device__ inline int pack_bf16(float a, float b) {
    union { int u; __bf16 h[2]; } p;
    p.h[0] = (__bf16)a;     // -> v_cvt_pk_bf16_f32
    p.h[1] = (__bf16)b;
    return p.u;
}

// ---------------------------------------------------------------------------
// Kernel 1: build A2 (64 x S64, bf16, zero-padded) and W1 (64 x S128, bf16) in ws.
// ---------------------------------------------------------------------------
__global__ __launch_bounds__(256) void prep_kernel(const float* __restrict__ tril,
                                                   const float* __restrict__ W1,
                                                   __bf16* __restrict__ ws_a2,
                                                   __bf16* __restrict__ ws_w1) {
    __shared__ float sA[NN * 64];
    __shared__ float sdinv[64];
    const int tid = threadIdx.x;

    // rebuild symmetric adjacency: W[i][j] = tril[hi*(hi+1)/2+lo]  (pow2 index math)
    for (int idx = tid; idx < NN * 64; idx += 256) {
        const int i = idx >> 6, j = idx & 63;
        if (j < NN) {
            const int hi = i > j ? i : j, lo = i > j ? j : i;
            sA[idx] = tril[((hi * (hi + 1)) >> 1) + lo];
        }
    }
    __syncthreads();

    if (tid < NN) {
        float s = 0.0f;
        for (int j = 0; j < NN; ++j) s += fabsf(sA[tid * 64 + j]);
        sdinv[tid] = (s > 0.0f) ? rsqrtf(s) : 0.0f;
    }
    __syncthreads();

    for (int idx = tid; idx < NN * 64; idx += 256) {
        const int i = idx >> 6, j = idx & 63;
        if (j < NN) sA[idx] *= sdinv[i] * sdinv[j];
    }
    __syncthreads();

    // A2 = A @ A, zero-padded to 64 x S64, bf16. Thread -> (row = tid>>2, cols tid&3 ::4)
    {
        const int i = tid >> 2;
        for (int j = (tid & 3); j < S64; j += 4) {
            float s = 0.0f;
            if (i < NN && j < NN)
                for (int m = 0; m < NN; ++m) s += sA[i * 64 + m] * sA[m * 64 + j];
            ws_a2[i * S64 + j] = (__bf16)s;
        }
    }
    // W1 in natural [n][k] layout (this IS the transposed-B layout), padded to S128
    {
        const int n = tid >> 2;
        for (int k = (tid & 3); k < S128; k += 4)
            ws_w1[n * S128 + k] = (k < FIN) ? (__bf16)W1[n * FIN + k] : (__bf16)0.0f;
    }
}

// ---------------------------------------------------------------------------
// Kernel 2: one block per graph. Fused A2@x -> relu(t@W1^T+b1) -> pool -> head.
// ---------------------------------------------------------------------------
__global__ __launch_bounds__(256) void graph_kernel(const float* __restrict__ x,
                                                    const float* __restrict__ b1,
                                                    const float* __restrict__ W2,
                                                    const float* __restrict__ b2,
                                                    const __bf16* __restrict__ ws_a2,
                                                    const __bf16* __restrict__ ws_w1,
                                                    float* __restrict__ out) {
    __shared__ __attribute__((aligned(16))) __bf16 s_a2[64 * S64];    //  9.0 KB, [m][k]
    __shared__ __attribute__((aligned(16))) __bf16 s_xT[FIN * S64];   // 18.0 KB, [n][k]
    __shared__ __attribute__((aligned(16))) __bf16 s_t[64 * S128];    // 17.0 KB, [m][k]
    __shared__ __attribute__((aligned(16))) __bf16 s_w1[HH * S128];   // 17.0 KB, [n][k]
    __shared__ float s_pooled[64];

    const int tid  = threadIdx.x;
    const int g    = blockIdx.x;
    const int lane = tid & 31;
    const int wave = tid >> 5;

    // ---- stage x_g: process node-row pairs (r, r+1) so each feature column gives
    // one packed-bf16 b32 store into the transposed tile s_xT[n][k].
    const float4* xg4 = (const float4*)(x + (size_t)g * NN * FIN);
    for (int idx = tid; idx < (NN / 2) * (FIN / 4); idx += 256) {
        const int r  = (idx >> 5) << 1;   // even node row
        const int c4 = idx & 31;          // feature col / 4
        const float4 v0 = xg4[(r + 0) * (FIN / 4) + c4];
        const float4 v1 = xg4[(r + 1) * (FIN / 4) + c4];
        const int c = c4 << 2;
        *(int*)&s_xT[(c + 0) * S64 + r] = pack_bf16(v0.x, v1.x);
        *(int*)&s_xT[(c + 1) * S64 + r] = pack_bf16(v0.y, v1.y);
        *(int*)&s_xT[(c + 2) * S64 + r] = pack_bf16(v0.z, v1.z);
        *(int*)&s_xT[(c + 3) * S64 + r] = pack_bf16(v0.w, v1.w);
    }
    if (tid < FIN) *(int*)&s_xT[tid * S64 + NN] = 0;   // zero-pad k = 62, 63

    // A2 and W1 (bf16, pre-padded in ws): straight int4 copies
    {
        const int4* src = (const int4*)ws_a2;
        int4* dst = (int4*)s_a2;
        for (int idx = tid; idx < 64 * S64 / 8; idx += 256) dst[idx] = src[idx];
    }
    {
        const int4* src = (const int4*)ws_w1;
        int4* dst = (int4*)s_w1;
        for (int idx = tid; idx < HH * S128 / 8; idx += 256) dst[idx] = src[idx];
    }
    if (tid < 64) s_pooled[tid] = 0.0f;
    __syncthreads();

    // ---- GEMM1: t[64,128] = A2[64,64] @ x[64,128] ----
    for (int tile = wave; tile < 32; tile += 8) {
        const int mt = tile >> 3, nt = tile & 7;
        const __bf16* arow = s_a2 + (mt * 16 + (lane & 15)) * S64;
        const __bf16* brow = s_xT + (nt * 16 + (lane & 15)) * S64;
        v8f c = {};
#pragma unroll
        for (int kc = 0; kc < 2; ++kc) {
            v16bf a = frag_row(arow + kc * 32, lane);
            v16bf b = frag_row(brow + kc * 32, lane);
            c = __builtin_amdgcn_wmma_f32_16x16x32_bf16(false, a, false, b,
                                                        (short)0, c, false, false);
        }
        const int col  = nt * 16 + (lane & 15);
        const int rowb = mt * 16 + ((lane >> 4) << 3);
#pragma unroll
        for (int r = 0; r < 8; ++r) s_t[(rowb + r) * S128 + col] = (__bf16)c[r];
    }
    __syncthreads();

    // ---- GEMM2: h[64,64] = relu(t[64,128] @ W1^T[128,64] + b1); pool cols ----
    for (int tile = wave; tile < 16; tile += 8) {
        const int mt   = tile >> 2, nt = tile & 3;
        const int ncol = nt * 16 + (lane & 15);
        const __bf16* arow = s_t + (mt * 16 + (lane & 15)) * S128;
        const __bf16* brow = s_w1 + ncol * S128;
        const float bias = b1[ncol];
        v8f c;
#pragma unroll
        for (int r = 0; r < 8; ++r) c[r] = bias;
#pragma unroll
        for (int kc = 0; kc < 4; ++kc) {
            v16bf a = frag_row(arow + kc * 32, lane);
            v16bf b = frag_row(brow + kc * 32, lane);
            c = __builtin_amdgcn_wmma_f32_16x16x32_bf16(false, a, false, b,
                                                        (short)0, c, false, false);
        }
        const int rowb = mt * 16 + ((lane >> 4) << 3);
        float acc = 0.0f;
#pragma unroll
        for (int r = 0; r < 8; ++r) {
            float h = c[r] > 0.0f ? c[r] : 0.0f;   // relu
            if (rowb + r < NN) acc += h;           // pool only real nodes
        }
        atomicAdd(&s_pooled[ncol], acc);           // ds_add_f32
    }
    __syncthreads();

    // ---- outputs: latent [B,64] then logp [B,3] ----
    if (tid < 64) out[(size_t)g * 64 + tid] = s_pooled[tid];
    if (tid == 0) {
        float l[3];
        float mx = -3.0e38f;
        for (int cc = 0; cc < 3; ++cc) {
            float s = b2[cc];
            for (int k = 0; k < 64; ++k) s += W2[cc * 64 + k] * s_pooled[k];
            l[cc] = s;
            mx = fmaxf(mx, s);
        }
        float se = 0.0f;
        for (int cc = 0; cc < 3; ++cc) se += __expf(l[cc] - mx);
        const float lse = __logf(se) + mx;
        float* lp = out + (size_t)NB * 64 + (size_t)g * 3;
        for (int cc = 0; cc < 3; ++cc) lp[cc] = l[cc] - lse;
    }
}

extern "C" void kernel_launch(void* const* d_in, const int* in_sizes, int n_in,
                              void* d_out, int out_size, void* d_ws, size_t ws_size,
                              hipStream_t stream) {
    // setup_inputs order:
    // 0:x 1:edge_weight_tril 2:W1 3:b1 4:W2 5:b2 6:edge_index 7:batch 8:batch_size
    const float* x    = (const float*)d_in[0];
    const float* tril = (const float*)d_in[1];
    const float* W1   = (const float*)d_in[2];
    const float* b1   = (const float*)d_in[3];
    const float* W2   = (const float*)d_in[4];
    const float* b2   = (const float*)d_in[5];

    __bf16* ws_a2 = (__bf16*)d_ws;                                        // 64*S64 bf16
    __bf16* ws_w1 = (__bf16*)((char*)d_ws + 64 * S64 * sizeof(__bf16));   // HH*S128 bf16

    prep_kernel<<<1, 256, 0, stream>>>(tril, W1, ws_a2, ws_w1);
    graph_kernel<<<NB, 256, 0, stream>>>(x, b1, W2, b2, ws_a2, ws_w1, (float*)d_out);
}
